// VoxelResBackBone8x_V2_16174846837209
// MI455X (gfx1250) — matrix operand order repaired and checked by reading, not compiled, not run on hardware
//
#include <hip/hip_runtime.h>

// ---------------------------------------------------------------------------
// VoxelResBackBone8x_V2 forward for MI455X (gfx1250, wave32, WMMA).
// All convs run as implicit-GEMM on v_wmma_f32_16x16x32_f16:
//   M = 16 output voxels along W, N = 16 output channels, K = taps*Cin.
// Conv kernel is templated on (CIN, KD): the K loop fully unrolls, tap
// decode and spatial offsets constant-fold, and each K step is just
// address-add + bounds-select + b128 load + v_wmma.
// BN (+bias) folded to per-channel affine; mask / residual / ReLU fused
// into the conv epilogue. All intermediates are f16 channel-last [D,H,W,C].
// ---------------------------------------------------------------------------

typedef __attribute__((ext_vector_type(16))) _Float16 v16h;
typedef __attribute__((ext_vector_type(8)))  _Float16 v8h;
typedef __attribute__((ext_vector_type(8)))  float    v8f;

#define CDIV(a, b) (((a) + (b) - 1) / (b))

// ---- weight pack: fp32 [Cout][Cin][kd][3][3] -> f16 B-fragment layout ------
// packed[(kt*nCT+ct)*32 + lane][j] = W[K = kt*32 + (j>>3)*16 + (lane>>4)*8 + (j&7)]
//                                     [cout = ct*16 + (lane&15)]  (K tap-major)
__global__ void pack_w_k(const float* __restrict__ w, _Float16* __restrict__ wpk,
                         int Cin, int Cout, int kd) {
  const int taps = kd * 9, Ktot = taps * Cin, nCT = Cout >> 4;
  const int nKT = (Ktot + 31) >> 5;
  const long total = (long)nKT * nCT * 32 * 16;
  long i = blockIdx.x * (long)blockDim.x + threadIdx.x;
  if (i >= total) return;
  const int j = (int)(i & 15);
  long t = i >> 4;
  const int lane = (int)(t & 31); t >>= 5;
  const int ct = (int)(t % nCT);
  const int kt = (int)(t / nCT);
  const int K = kt * 32 + (j >> 3) * 16 + (lane >> 4) * 8 + (j & 7);
  const int cout = ct * 16 + (lane & 15);
  _Float16 v = (_Float16)0.f;
  if (K < Ktot) {
    const int tap = K / Cin;
    const int cin = K - tap * Cin;
    v = (_Float16)w[((size_t)cout * Cin + cin) * taps + tap];
  }
  wpk[i] = v;
}

// ---- BN(+bias) -> per-channel affine y = a*x + b ---------------------------
__global__ void make_affine_k(const float* __restrict__ g, const float* __restrict__ bta,
                              const float* __restrict__ mu, const float* __restrict__ var,
                              const float* __restrict__ bias,
                              float* __restrict__ A, float* __restrict__ B, int C) {
  int c = blockIdx.x * blockDim.x + threadIdx.x;
  if (c >= C) return;
  float a = g[c] * rsqrtf(var[c] + 1e-3f);
  float b0 = bias ? bias[c] : 0.f;
  A[c] = a;
  B[c] = (b0 - mu[c]) * a + bta[c];
}

// ---- implicit-GEMM conv, templated on (CIN, KD); stride 1/2, pad 1 (xy) ----
template <int CIN, int KD>
__global__ __launch_bounds__(128)
void conv_wmma_t(const _Float16* __restrict__ act, const _Float16* __restrict__ wpk,
                 const float* __restrict__ affA, const float* __restrict__ affB,
                 const _Float16* __restrict__ msk, const _Float16* __restrict__ res,
                 _Float16* __restrict__ out,
                 int Din, int Hin, int Win,
                 int Dout, int Hout, int Wout, int Cout,
                 int stride, int relu) {
  constexpr int TAPS = KD * 9;
  constexpr int KTOT = TAPS * CIN;
  constexpr int NKT  = (KTOT + 31) / 32;

  const int lane = threadIdx.x & 31;
  const int nCT = Cout >> 4;
  const int wT  = Wout >> 4;
  const long nWaves = (long)Dout * Hout * wT * nCT;
  long gw = (long)blockIdx.x * (blockDim.x >> 5) + (threadIdx.x >> 5);
  if (gw >= nWaves) return;                 // uniform per wave: EXEC stays all-1s
  const int ct = (int)(gw % nCT);
  long tile = gw / nCT;
  const int xt = (int)(tile % wT); tile /= wT;
  const int yo = (int)(tile % Hout);
  const int zo = (int)(tile / Hout);

  const int mrow  = lane & 15;              // A row = output voxel within tile
  const int khalf = lane >> 4;              // K-half selector (ISA A layout)
  const int ibx = (xt * 16 + mrow) * stride - 1;
  const int iby = yo * stride - 1;
  const int ibz = zo * stride - (KD == 3 ? 1 : 0);

  const size_t bstep = (size_t)nCT * 32 * 16;
  const _Float16* bbase = wpk + ((size_t)ct * 32 + lane) * 16;

  v8f acc = {};
#pragma unroll
  for (int kt = 0; kt < NKT; ++kt) {
    v16h a;
#pragma unroll
    for (int gsel = 0; gsel < 2; ++gsel) {
      const int base = kt * 32 + gsel * 16;         // constant after unroll
      v8h av = {};
      if (base < KTOT) {                            // folds away on padded tail
        if (CIN >= 16) {
          // whole 8-half group stays inside one tap; tap decode is constant
          const int tap  = base / CIN;
          const int kz = tap / 9, r9 = tap % 9;
          const int ky = r9 / 3, kx = r9 % 3;
          const int cin0 = (base % CIN) + khalf * 8;
          const int iz = ibz + kz, iy = iby + ky, ix = ibx + kx;
          if ((unsigned)iz < (unsigned)Din && (unsigned)iy < (unsigned)Hin &&
              (unsigned)ix < (unsigned)Win) {
            const _Float16* p =
                act + ((size_t)((size_t)(iz * Hin + iy) * Win + ix)) * CIN + cin0;
            av = *(const v8h*)p;                    // 16B contiguous load
          }
        } else {
          // CIN == 4 (input conv): each 8-half group = two taps, 4 halfs each
#pragma unroll
          for (int h4 = 0; h4 < 2; ++h4) {
            const int tap = base / 4 + khalf * 2 + h4;   // tiny runtime decode
            if (tap < TAPS) {
              const int kz = tap / 9, r9 = tap % 9;
              const int ky = r9 / 3, kx = r9 % 3;
              const int iz = ibz + kz, iy = iby + ky, ix = ibx + kx;
              if ((unsigned)iz < (unsigned)Din && (unsigned)iy < (unsigned)Hin &&
                  (unsigned)ix < (unsigned)Win) {
                const _Float16* p =
                    act + ((size_t)((size_t)(iz * Hin + iy) * Win + ix)) * 4;
#pragma unroll
                for (int j = 0; j < 4; ++j) av[h4 * 4 + j] = p[j];
              }
            }
          }
        }
      }
#pragma unroll
      for (int j = 0; j < 8; ++j) a[gsel * 8 + j] = av[j];
    }
    const v16h b = *(const v16h*)(bbase + (size_t)kt * bstep);  // pre-packed frag
    acc = __builtin_amdgcn_wmma_f32_16x16x32_f16(false, a, false, b,
                                                 (short)0, acc, false, false);
  }

  // epilogue: affine -> mask -> residual -> relu (D layout: N=lane&15, M=r+8*khalf)
  const int co = ct * 16 + (lane & 15);
  const float sa = affA[co];
  const float sb = affB[co];
#pragma unroll
  for (int r = 0; r < 8; ++r) {
    const int Mr = r + khalf * 8;
    const size_t vox = ((size_t)zo * Hout + yo) * (size_t)Wout + (xt * 16 + Mr);
    float v = acc[r] * sa + sb;
    if (msk) v *= (float)msk[vox];
    if (res) v += (float)res[vox * Cout + co];
    if (relu) v = fmaxf(v, 0.f);
    out[vox * Cout + co] = (_Float16)v;
  }
}

// ---- input: fp32 NCDHW (C=4) * mask -> f16 [vox][4] + f16 mask -------------
__global__ void cvt_input_k(const float* __restrict__ x, const unsigned char* __restrict__ mk,
                            _Float16* __restrict__ act, _Float16* __restrict__ m,
                            long nV, int Cin) {
  long v = blockIdx.x * (long)blockDim.x + threadIdx.x;
  if (v >= nV) return;
  float mm = mk[v] ? 1.f : 0.f;
  m[v] = (_Float16)mm;
  for (int c = 0; c < Cin; ++c) act[v * Cin + c] = (_Float16)(x[(long)c * nV + v] * mm);
}

// ---- strided 3^3 window "any active" mask (SparseConv3d output mask) -------
__global__ void mask_down_k(const _Float16* __restrict__ mi, _Float16* __restrict__ mo,
                            int Di, int Hi, int Wi, int Do, int Ho, int Wo) {
  long i = blockIdx.x * (long)blockDim.x + threadIdx.x;
  long n = (long)Do * Ho * Wo;
  if (i >= n) return;
  int x = (int)(i % Wo); long t = i / Wo;
  int y = (int)(t % Ho); int z = (int)(t / Ho);
  float any = 0.f;
  for (int dz = 0; dz < 3; ++dz) {
    int iz = 2 * z + dz - 1; if ((unsigned)iz >= (unsigned)Di) continue;
    for (int dy = 0; dy < 3; ++dy) {
      int iy = 2 * y + dy - 1; if ((unsigned)iy >= (unsigned)Hi) continue;
      for (int dx = 0; dx < 3; ++dx) {
        int ix = 2 * x + dx - 1; if ((unsigned)ix >= (unsigned)Wi) continue;
        if ((float)mi[((size_t)(iz * Hi + iy)) * Wi + ix] > 0.f) any = 1.f;
      }
    }
  }
  mo[i] = (_Float16)any;
}

// ---- channel attention: f *= sigmoid(2 * W2 relu(W1 f)) --------------------
__global__ void chan_att_k(_Float16* __restrict__ f, const float* __restrict__ w1,
                           const float* __restrict__ w2, int C, int R, long nV) {
  long v = blockIdx.x * (long)blockDim.x + threadIdx.x;
  if (v >= nV) return;
  float h[8];
  for (int r = 0; r < R; ++r) {
    float s = 0.f;
    for (int c = 0; c < C; ++c) s += w1[r * C + c] * (float)f[v * C + c];
    h[r] = fmaxf(s, 0.f);
  }
  for (int c = 0; c < C; ++c) {
    float y = 0.f;
    for (int r = 0; r < R; ++r) y += w2[c * R + r] * h[r];
    float sg = 1.f / (1.f + __expf(-2.f * y));
    f[v * C + c] = (_Float16)((float)f[v * C + c] * sg);
  }
}

// ---- per-voxel channel mean/max -> s[vox][2] -------------------------------
__global__ void spat_stats_k(const _Float16* __restrict__ f, _Float16* __restrict__ s,
                             int C, long nV) {
  long v = blockIdx.x * (long)blockDim.x + threadIdx.x;
  if (v >= nV) return;
  float sum = 0.f, mx = -3.0e38f;
  for (int c = 0; c < C; ++c) {
    float t = (float)f[v * C + c];
    sum += t;
    mx = fmaxf(mx, t);
  }
  s[v * 2]     = (_Float16)(sum / (float)C);
  s[v * 2 + 1] = (_Float16)mx;
}

// ---- spatial attention: f *= sigmoid(conv_{2->1}(s)) -----------------------
__global__ void spat_att_k(_Float16* __restrict__ f, const _Float16* __restrict__ s,
                           const float* __restrict__ w, int D, int H, int W, int C, int kd) {
  long nV = (long)D * H * W;
  long v = blockIdx.x * (long)blockDim.x + threadIdx.x;
  if (v >= nV) return;
  int x = (int)(v % W); long t = v / W;
  int y = (int)(t % H); int z = (int)(t / H);
  const int taps = kd * 9;
  float acc = 0.f;
  for (int tap = 0; tap < taps; ++tap) {
    int kz = tap / 9, r9 = tap % 9, ky = r9 / 3, kx = r9 % 3;
    int iz = z + kz - (kd == 3 ? 1 : 0), iy = y + ky - 1, ix = x + kx - 1;
    if ((unsigned)iz < (unsigned)D && (unsigned)iy < (unsigned)H &&
        (unsigned)ix < (unsigned)W) {
      long vi = ((long)iz * H + iy) * W + ix;
      acc += w[tap] * (float)s[vi * 2] + w[taps + tap] * (float)s[vi * 2 + 1];
    }
  }
  float sg = 1.f / (1.f + __expf(-acc));
  for (int c = 0; c < C; ++c) f[v * C + c] = (_Float16)((float)f[v * C + c] * sg);
}

// ---- BEV: sum over z -------------------------------------------------------
__global__ void bev_sum_k(const _Float16* __restrict__ f, _Float16* __restrict__ bev,
                          int D, int H, int W, int C) {
  long n = (long)H * W * C;
  long i = blockIdx.x * (long)blockDim.x + threadIdx.x;
  if (i >= n) return;
  int c = (int)(i % C);
  long t = i / C;                       // y*W + x
  float s = 0.f;
  for (int z = 0; z < D; ++z) s += (float)f[((long)z * H * W + t) * C + c];
  bev[i] = (_Float16)s;
}

// ---- BEV mask: dilate_3x3(any_z(mask)) -------------------------------------
__global__ void bev_mask_k(const _Float16* __restrict__ m4, _Float16* __restrict__ mo,
                           int D, int H, int W) {
  int i = blockIdx.x * blockDim.x + threadIdx.x;
  if (i >= H * W) return;
  int x = i % W, y = i / W;
  float any = 0.f;
  for (int dy = -1; dy <= 1; ++dy) {
    int yy = y + dy; if ((unsigned)yy >= (unsigned)H) continue;
    for (int dx = -1; dx <= 1; ++dx) {
      int xx = x + dx; if ((unsigned)xx >= (unsigned)W) continue;
      for (int z = 0; z < D; ++z)
        if ((float)m4[((long)z * H + yy) * W + xx] > 0.f) any = 1.f;
    }
  }
  mo[i] = (_Float16)any;
}

// ---- f16 [H,W,C] -> fp32 NCHW output ---------------------------------------
__global__ void store_out_k(const _Float16* __restrict__ g, float* __restrict__ o,
                            int H, int W, int C) {
  long n = (long)C * H * W;
  long i = blockIdx.x * (long)blockDim.x + threadIdx.x;
  if (i >= n) return;
  int c = (int)(i / ((long)H * W));
  long rem = i - (long)c * H * W;
  o[i] = (float)g[rem * C + c];
}

// ---------------------------------------------------------------------------
// host-side orchestration helpers
// ---------------------------------------------------------------------------
static void run_conv(hipStream_t st, const _Float16* act, const void* w,
                     const void* bnb, const void* bng, const void* bnm, const void* bnv,
                     const void* bias, const _Float16* msk, const _Float16* res, int relu,
                     _Float16* out, _Float16* wpk, float* affA, float* affB,
                     int Di, int Hi, int Wi, int Cin,
                     int Do, int Ho, int Wo, int Cout, int stride, int kd) {
  const int taps = kd * 9, Ktot = taps * Cin;
  const int nKT = (Ktot + 31) / 32, nCT = Cout / 16;
  const long pk = (long)nKT * nCT * 32 * 16;
  pack_w_k<<<(unsigned)CDIV(pk, 256), 256, 0, st>>>((const float*)w, wpk, Cin, Cout, kd);
  make_affine_k<<<(unsigned)CDIV(Cout, 64), 64, 0, st>>>(
      (const float*)bng, (const float*)bnb, (const float*)bnm, (const float*)bnv,
      (const float*)bias, affA, affB, Cout);
  const long nWaves = (long)Do * Ho * (Wo / 16) * nCT;
  const unsigned blks = (unsigned)CDIV(nWaves, 4);
#define LAUNCH_CONV(CI, KDV)                                                   \
  conv_wmma_t<CI, KDV><<<blks, 128, 0, st>>>(act, wpk, affA, affB, msk, res,   \
                                             out, Di, Hi, Wi, Do, Ho, Wo,      \
                                             Cout, stride, relu)
  if (kd == 3) {
    if (Cin == 4)        LAUNCH_CONV(4, 3);
    else if (Cin == 16)  LAUNCH_CONV(16, 3);
    else if (Cin == 32)  LAUNCH_CONV(32, 3);
    else if (Cin == 64)  LAUNCH_CONV(64, 3);
    else                 LAUNCH_CONV(128, 3);
  } else {
    LAUNCH_CONV(128, 1);
  }
#undef LAUNCH_CONV
}

// SparseBasicBlock: conv1(+b1,bn1,mask,relu) -> conv2(+b2,bn2,mask,+f,relu), in place on f
static void run_block(hipStream_t st, void* const* din, int bi, _Float16* f, _Float16* t,
                      const _Float16* m, _Float16* wpk, float* aA, float* aB,
                      int C, int D, int H, int W) {
  run_conv(st, f, din[bi + 10], din[bi + 2], din[bi + 3], din[bi + 4], din[bi + 5],
           din[bi + 0], m, nullptr, 1, t, wpk, aA, aB, D, H, W, C, D, H, W, C, 1, 3);
  run_conv(st, t, din[bi + 11], din[bi + 6], din[bi + 7], din[bi + 8], din[bi + 9],
           din[bi + 1], m, f, 1, f, wpk, aA, aB, D, H, W, C, D, H, W, C, 1, 3);
}

static void run_attn(hipStream_t st, void* const* din, int iw1, int iw2, int isa,
                     _Float16* f, _Float16* stats, int C, int R,
                     int D, int H, int W, int kd) {
  long nv = (long)D * H * W;
  chan_att_k<<<(unsigned)CDIV(nv, 128), 128, 0, st>>>(
      f, (const float*)din[iw1], (const float*)din[iw2], C, R, nv);
  spat_stats_k<<<(unsigned)CDIV(nv, 128), 128, 0, st>>>(f, stats, C, nv);
  spat_att_k<<<(unsigned)CDIV(nv, 128), 128, 0, st>>>(
      f, stats, (const float*)din[isa], D, H, W, C, kd);
}

// d_in index map: insertion order (x, mask, params); params flattened
// depth-first with alphabetical dict keys (jax-pytree convention):
//  0:x 1:mask 2:ca2d.w1 3:ca2d.w2 4-7:inp.bn{b,g,m,v} 8:inp.w
//  9-12:out.bn{b,g,m,v} 13:out.w
//  14:s1 (blk: +0 b1,+1 b2,+2..5 bn1{b,g,m,v},+6..9 bn2,+10 w1,+11 w2;
//         blk1=base, blk2=base+12, ca.w1=+24, ca.w2=+25,
//         s1: sa.w=+26 | s2-4: down.bn{b,g,m,v}=+26..29, down.w=+30, sa.w=+31)
//  41:s2  73:s3  105:s4  137:sa2d.w
//  138:shared.b 139-142:shared.bn{b,g,m,v} 143:shared.w
extern "C" void kernel_launch(void* const* d_in, const int* in_sizes, int n_in,
                              void* d_out, int out_size, void* d_ws, size_t ws_size,
                              hipStream_t stream) {
  (void)in_sizes; (void)n_in; (void)out_size; (void)ws_size;
  const int D1 = 25, H1 = 256, W1 = 256;
  const int D2 = 13, H2 = 128, W2 = 128;
  const int D3 = 7,  H3 = 64,  W3 = 64;
  const int D4 = 4,  H4 = 32,  W4 = 32;
  const long V1 = (long)D1 * H1 * W1, V2 = (long)D2 * H2 * W2;
  const long V3 = (long)D3 * H3 * W3, V4 = (long)D4 * H4 * W4;

  char* pp = (char*)d_ws;
  auto alloc = [&](size_t b) -> char* {
    char* r = pp;
    pp += (b + 255) & ~(size_t)255;
    return r;
  };
  _Float16* bufA  = (_Float16*)alloc((size_t)V1 * 16 * 2);
  _Float16* bufB  = (_Float16*)alloc((size_t)V1 * 16 * 2);
  _Float16* m1    = (_Float16*)alloc((size_t)V1 * 2);
  _Float16* m2    = (_Float16*)alloc((size_t)V2 * 2);
  _Float16* m3    = (_Float16*)alloc((size_t)V3 * 2);
  _Float16* m4    = (_Float16*)alloc((size_t)V4 * 2);
  _Float16* stats = (_Float16*)alloc((size_t)V1 * 2 * 2);
  _Float16* wpk   = (_Float16*)alloc((size_t)2 << 20);
  float*    affA  = (float*)alloc(512);
  float*    affB  = (float*)alloc(512);
  _Float16* bev   = (_Float16*)alloc((size_t)H4 * W4 * 128 * 2);
  _Float16* g2    = (_Float16*)alloc((size_t)H4 * W4 * 128 * 2);
  _Float16* mo    = (_Float16*)alloc((size_t)H4 * W4 * 2);

  // input: x -> channel-last f16 + mask
  cvt_input_k<<<(unsigned)CDIV(V1, 256), 256, 0, stream>>>(
      (const float*)d_in[0], (const unsigned char*)d_in[1], bufA, m1, V1, 4);

  // input conv 4->16, bn, relu, mask
  run_conv(stream, bufA, d_in[8], d_in[4], d_in[5], d_in[6], d_in[7], nullptr,
           m1, nullptr, 1, bufB, wpk, affA, affB, D1, H1, W1, 4, D1, H1, W1, 16, 1, 3);
  _Float16* f = bufB;
  _Float16* t = bufA;

  // stage 1 @16ch, full res
  run_block(stream, d_in, 14, f, t, m1, wpk, affA, affB, 16, D1, H1, W1);
  run_block(stream, d_in, 26, f, t, m1, wpk, affA, affB, 16, D1, H1, W1);
  run_attn(stream, d_in, 38, 39, 40, f, stats, 16, 1, D1, H1, W1, 3);

  // stage 2: down 16->32 s2, then blocks @32ch
  mask_down_k<<<(unsigned)CDIV(V2, 256), 256, 0, stream>>>(m1, m2, D1, H1, W1, D2, H2, W2);
  run_conv(stream, f, d_in[41 + 30], d_in[41 + 26], d_in[41 + 27], d_in[41 + 28],
           d_in[41 + 29], nullptr, m2, nullptr, 1, t, wpk, affA, affB,
           D1, H1, W1, 16, D2, H2, W2, 32, 2, 3);
  { _Float16* x = f; f = t; t = x; }
  run_block(stream, d_in, 41, f, t, m2, wpk, affA, affB, 32, D2, H2, W2);
  run_block(stream, d_in, 53, f, t, m2, wpk, affA, affB, 32, D2, H2, W2);
  run_attn(stream, d_in, 65, 66, 72, f, stats, 32, 2, D2, H2, W2, 3);

  // stage 3: down 32->64 s2, then blocks @64ch
  mask_down_k<<<(unsigned)CDIV(V3, 256), 256, 0, stream>>>(m2, m3, D2, H2, W2, D3, H3, W3);
  run_conv(stream, f, d_in[73 + 30], d_in[73 + 26], d_in[73 + 27], d_in[73 + 28],
           d_in[73 + 29], nullptr, m3, nullptr, 1, t, wpk, affA, affB,
           D2, H2, W2, 32, D3, H3, W3, 64, 2, 3);
  { _Float16* x = f; f = t; t = x; }
  run_block(stream, d_in, 73, f, t, m3, wpk, affA, affB, 64, D3, H3, W3);
  run_block(stream, d_in, 85, f, t, m3, wpk, affA, affB, 64, D3, H3, W3);
  run_attn(stream, d_in, 97, 98, 104, f, stats, 64, 4, D3, H3, W3, 3);

  // stage 4: down 64->128 s2, then blocks @128ch
  mask_down_k<<<(unsigned)CDIV(V4, 256), 256, 0, stream>>>(m3, m4, D3, H3, W3, D4, H4, W4);
  run_conv(stream, f, d_in[105 + 30], d_in[105 + 26], d_in[105 + 27], d_in[105 + 28],
           d_in[105 + 29], nullptr, m4, nullptr, 1, t, wpk, affA, affB,
           D3, H3, W3, 64, D4, H4, W4, 128, 2, 3);
  { _Float16* x = f; f = t; t = x; }
  run_block(stream, d_in, 105, f, t, m4, wpk, affA, affB, 128, D4, H4, W4);
  run_block(stream, d_in, 117, f, t, m4, wpk, affA, affB, 128, D4, H4, W4);
  run_attn(stream, d_in, 129, 130, 136, f, stats, 128, 8, D4, H4, W4, 3);

  // BEV collapse + 2D masks
  bev_sum_k<<<(unsigned)CDIV((long)H4 * W4 * 128, 256), 256, 0, stream>>>(
      f, bev, D4, H4, W4, 128);
  bev_mask_k<<<(unsigned)CDIV(H4 * W4, 128), 128, 0, stream>>>(m4, mo, D4, H4, W4);

  // 2D conv 128->128 (bn, relu, mask), attention, shared conv (+bias, bn, relu, mask)
  _Float16* g1 = t;
  run_conv(stream, bev, d_in[13], d_in[9], d_in[10], d_in[11], d_in[12], nullptr,
           mo, nullptr, 1, g1, wpk, affA, affB, 1, H4, W4, 128, 1, H4, W4, 128, 1, 1);
  run_attn(stream, d_in, 2, 3, 137, g1, stats, 128, 8, 1, H4, W4, 1);
  run_conv(stream, g1, d_in[143], d_in[139], d_in[140], d_in[141], d_in[142], d_in[138],
           mo, nullptr, 1, g2, wpk, affA, affB, 1, H4, W4, 128, 1, H4, W4, 128, 1, 1);

  // fp32 NCHW output
  store_out_k<<<(unsigned)CDIV((long)128 * H4 * W4, 256), 256, 0, stream>>>(
      g2, (float*)d_out, H4, W4, 128);
}
